// GNN_13821204758889
// MI455X (gfx1250) — compile-verified, matrix operand order checked by reading, not run on hardware
//
#include <hip/hip_runtime.h>
#include <hip/hip_bf16.h>
#include <stdint.h>

// ---------------------------------------------------------------------------
// 2-layer GCN for MI455X (gfx1250, wave32).
//   h1  = X @ W1                         (WMMA f16 -> f32)
//   agg = scatter-add over edges of h1[src] * dinv[src]*dinv[dst]
//   h   = relu(agg + dinv^2 * h1 + b1)
//   h2  = h @ W2                         (WMMA f16 -> f32)
//   out = scatter(h2) + dinv^2 * h2 + b2
// ---------------------------------------------------------------------------

typedef __attribute__((ext_vector_type(16))) _Float16 v16h;
typedef __attribute__((ext_vector_type(8)))  float    v8f;

// ---------------------------------------------------------------------------
// Utility: fill a float buffer with a constant (used for deg=1.0 and agg=0).
// ---------------------------------------------------------------------------
__global__ void fill_kernel(float* __restrict__ p, long long n, float v) {
    long long i = (long long)blockIdx.x * blockDim.x + threadIdx.x;
    if (i < n) p[i] = v;
}

// ---------------------------------------------------------------------------
// Degree accumulation: deg[dst] += 1 per edge (deg pre-filled with 1.0 for
// the self loop).  edge_index is int64, laid out [2, E] row-major.
// ---------------------------------------------------------------------------
__global__ void degree_kernel(const long long* __restrict__ ei, long long E,
                              float* __restrict__ deg) {
    long long e = (long long)blockIdx.x * blockDim.x + threadIdx.x;
    if (e < E) {
        int d = (int)ei[E + e];
        atomicAdd(&deg[d], 1.0f);
    }
}

// dinv[i] = rsqrt(deg[i]), in place  (emits v_rsq_f32)
__global__ void rsqrt_kernel(float* __restrict__ p, long long n) {
    long long i = (long long)blockIdx.x * blockDim.x + threadIdx.x;
    if (i < n) p[i] = __frsqrt_rn(p[i]);
}

// ---------------------------------------------------------------------------
// WMMA GEMM:  H[nRows x NF] = X[nRows x K] * W[K x NF]
//
//   - W is pre-swizzled into LDS in the *exact* CDNA5 16-bit B-fragment
//     layout, so each lane fetches its whole 16-half B operand with a single
//     aligned 32-byte LDS load (one ds_load_b128 pair) per WMMA:
//        fragment index = (kTile*2 + kHalf)*NF + nCol, element j
//        kin = k % 32:  kin<16  -> kHalf = kin>>3,      j = kin&7
//                       kin>=16 -> kHalf = (kin-16)>>3, j = 8 + (kin&7)
//   - 256 threads = 8 waves per block.  Each wave owns one 16x16 C tile:
//       NTILES = NF/16 tiles across N, MSUB = 8/NTILES tiles down M.
//   - A fragment (16x32 f16) per the CDNA5 layout: lane = {kHalf = lane>>4,
//     m = lane&15}; halves hold K = kHalf*8 + [0..7], then 16+kHalf*8+[0..7].
//     Loaded as two contiguous float4 pairs from global, cvt_pk to f16.
//   - C layout: VGPR r, lane l -> row = (l>>4)*8 + r, col = l&15.
//   - Rows clamped (EXEC must stay all-ones for WMMA); stores guarded.
// ---------------------------------------------------------------------------
template <int K, int NF>
__global__ void gemm_wmma_kernel(const float* __restrict__ X,
                                 const float* __restrict__ W,
                                 float* __restrict__ H, int nRows) {
    constexpr int NTILES = NF / 16;
    constexpr int MSUB   = 8 / NTILES;          // M tiles per block
    constexpr int KT     = K / 32;              // k-tiles

    // B fragments, v16h-typed for guaranteed 32B alignment -> ds_load_b128.
    __shared__ v16h ldsB[KT * 2 * NF];

    // Cooperative swizzled W load (f32 -> f16, row-major -> fragment-major).
    {
        _Float16* lw = (_Float16*)ldsB;
        for (int i = threadIdx.x; i < K * NF; i += 256) {
            int k = i / NF, n = i % NF;
            int kTile = k >> 5;
            int kin   = k & 31;
            int kHalf = (kin & 15) >> 3;
            int j     = (kin & 7) + ((kin & 16) ? 8 : 0);
            lw[(((kTile * 2 + kHalf) * NF) + n) * 16 + j] = (_Float16)W[i];
        }
    }
    __syncthreads();

    const int wave  = threadIdx.x >> 5;
    const int lane  = threadIdx.x & 31;
    const int mSub  = wave / NTILES;
    const int nTile = wave % NTILES;

    const int mBase = (blockIdx.x * MSUB + mSub) * 16;
    const int kHalf = lane >> 4;                 // 0: K 0-7/16-23, 1: K 8-15/24-31
    const int mRow  = mBase + (lane & 15);
    const int rA    = (mRow < nRows) ? mRow : (nRows - 1);   // clamp, keep EXEC full
    const int nCol  = nTile * 16 + (lane & 15);

    const float* __restrict__ xrow = X + (size_t)rA * K;
    const v16h* __restrict__ bfrag = ldsB + kHalf * NF + nCol;

    v8f acc = {};

#pragma unroll
    for (int kTile = 0; kTile < KT; ++kTile) {
        const int k0 = kTile * 32;

        // ---- A fragment: two contiguous 8-float runs, converted to f16 ----
        const float4 f0 = *(const float4*)(xrow + k0 + kHalf * 8);
        const float4 f1 = *(const float4*)(xrow + k0 + kHalf * 8 + 4);
        const float4 g0 = *(const float4*)(xrow + k0 + 16 + kHalf * 8);
        const float4 g1 = *(const float4*)(xrow + k0 + 16 + kHalf * 8 + 4);
        v16h a;
        a[0]  = (_Float16)f0.x; a[1]  = (_Float16)f0.y;
        a[2]  = (_Float16)f0.z; a[3]  = (_Float16)f0.w;
        a[4]  = (_Float16)f1.x; a[5]  = (_Float16)f1.y;
        a[6]  = (_Float16)f1.z; a[7]  = (_Float16)f1.w;
        a[8]  = (_Float16)g0.x; a[9]  = (_Float16)g0.y;
        a[10] = (_Float16)g0.z; a[11] = (_Float16)g0.w;
        a[12] = (_Float16)g1.x; a[13] = (_Float16)g1.y;
        a[14] = (_Float16)g1.z; a[15] = (_Float16)g1.w;

        // ---- B fragment: one aligned 32-byte LDS load ----
        const v16h b = bfrag[kTile * 2 * NF];

        acc = __builtin_amdgcn_wmma_f32_16x16x32_f16(
            /*neg_a=*/false, a, /*neg_b=*/false, b,
            /*c_mod=*/(short)0, acc, /*reuse_a=*/false, /*reuse_b=*/false);
    }

    // ---- C store per the 16x16 f32 C layout ----
#pragma unroll
    for (int r = 0; r < 8; ++r) {
        int row = mBase + kHalf * 8 + r;
        if (row < nRows)
            H[(size_t)row * NF + nCol] = acc[r];
    }
}

// ---------------------------------------------------------------------------
// Edge scatter:  agg[dst] += h[src] * dinv[src]*dinv[dst]
// One edge handled by F/4 consecutive threads, float4 gather + 4 atomics.
// agg fits in the 192MB L2 so atomics stay L2-resident.
// ---------------------------------------------------------------------------
template <int F>
__global__ void scatter_kernel(const long long* __restrict__ ei, long long E,
                               const float* __restrict__ h,
                               const float* __restrict__ dinv,
                               float* __restrict__ agg) {
    constexpr int TPE = F / 4;                   // threads per edge
    long long t = (long long)blockIdx.x * blockDim.x + threadIdx.x;
    long long e = t / TPE;
    if (e >= E) return;
    int fi = (int)(t % TPE) * 4;

    int s = (int)ei[e];
    int d = (int)ei[E + e];
    float nrm = dinv[s] * dinv[d];

    const float4 v = *(const float4*)(h + (size_t)s * F + fi);
    float* base = agg + (size_t)d * F + fi;
    atomicAdd(base + 0, v.x * nrm);
    atomicAdd(base + 1, v.y * nrm);
    atomicAdd(base + 2, v.z * nrm);
    atomicAdd(base + 3, v.w * nrm);
}

// ---------------------------------------------------------------------------
// Self-loop + bias (+ optional ReLU), in place on agg:
//   agg[i,f] = act(agg[i,f] + dinv[i]^2 * h[i,f] + bias[f])
// ---------------------------------------------------------------------------
template <int F, bool RELU>
__global__ void selfloop_kernel(float* __restrict__ agg,
                                const float* __restrict__ h,
                                const float* __restrict__ dinv,
                                const float* __restrict__ bias, long long n) {
    long long t = (long long)blockIdx.x * blockDim.x + threadIdx.x;
    if (t >= n) return;
    long long i = t / F;
    int f = (int)(t % F);
    float di = dinv[i];
    float v = agg[t] + di * di * h[t] + bias[f];
    if (RELU) v = v > 0.0f ? v : 0.0f;
    agg[t] = v;
}

// ---------------------------------------------------------------------------
// Launch
// ---------------------------------------------------------------------------
extern "C" void kernel_launch(void* const* d_in, const int* in_sizes, int n_in,
                              void* d_out, int out_size, void* d_ws, size_t ws_size,
                              hipStream_t stream) {
    const float*     x  = (const float*)d_in[0];           // [N,256]
    const long long* ei = (const long long*)d_in[1];       // [2,E] int64
    const float*     W1 = (const float*)d_in[2];           // [256,64]
    const float*     b1 = (const float*)d_in[3];           // [64]
    const float*     W2 = (const float*)d_in[4];           // [64,16]
    const float*     b2 = (const float*)d_in[5];           // [16]
    float*           out = (float*)d_out;                  // [N,16]

    const long long N = in_sizes[0] / 256;
    const long long E = in_sizes[1] / 2;

    // Scratch layout (floats): dinv[N] | h1[N*64] | hbuf[N*64] | h2[N*16]
    float* dinv = (float*)d_ws;
    float* h1   = dinv + N;
    float* hbuf = h1 + N * 64;
    float* h2   = hbuf + N * 64;

    const int T = 256;
    auto blocks = [](long long work, int t) -> unsigned {
        return (unsigned)((work + t - 1) / t);
    };

    // ---- degrees + rsqrt normalization ----
    fill_kernel<<<blocks(N, T), T, 0, stream>>>(dinv, N, 1.0f);        // +1 self loop
    degree_kernel<<<blocks(E, T), T, 0, stream>>>(ei, E, dinv);
    rsqrt_kernel<<<blocks(N, T), T, 0, stream>>>(dinv, N);

    // ---- layer 1: h1 = X @ W1  (block covers 32 rows x 64 cols) ----
    gemm_wmma_kernel<256, 64><<<blocks(N, 32), T, 0, stream>>>(x, W1, h1, (int)N);

    // ---- layer 1 aggregation ----
    fill_kernel<<<blocks(N * 64, T), T, 0, stream>>>(hbuf, N * 64, 0.0f);
    scatter_kernel<64><<<blocks(E * 16, T), T, 0, stream>>>(ei, E, h1, dinv, hbuf);
    selfloop_kernel<64, true><<<blocks(N * 64, T), T, 0, stream>>>(hbuf, h1, dinv, b1, N * 64);

    // ---- layer 2: h2 = h @ W2  (block covers 128 rows x 16 cols) ----
    gemm_wmma_kernel<64, 16><<<blocks(N, 128), T, 0, stream>>>(hbuf, W2, h2, (int)N);

    // ---- layer 2 aggregation straight into d_out ----
    fill_kernel<<<blocks(N * 16, T), T, 0, stream>>>(out, N * 16, 0.0f);
    scatter_kernel<16><<<blocks(E * 4, T), T, 0, stream>>>(ei, E, h2, dinv, out);
    selfloop_kernel<16, false><<<blocks(N * 16, T), T, 0, stream>>>(out, h2, dinv, b2, N * 16);
}
